// BERT_LSTM_2508260901331
// MI455X (gfx1250) — matrix operand order repaired and checked by reading, not compiled, not run on hardware
//
#include <hip/hip_runtime.h>
#include <cstdint>
#include <cstddef>

// Problem constants (match reference)
#define E_    768
#define H_    1024
#define HD_   512
#define V_    256
#define NOUT_ 2
#define B_    64
#define S_    512
#define G4_   (4 * H_)   // 4096

typedef __bf16 bf16_t;
typedef __attribute__((ext_vector_type(8)))  bf16_t v8bf;
typedef __attribute__((ext_vector_type(16))) bf16_t v16bf;
typedef __attribute__((ext_vector_type(8)))  float  v8f;

// ---------- helpers ----------
__device__ __forceinline__ unsigned short f2bf(float f) {
    unsigned int u = __float_as_uint(f);
    u += 0x7FFFu + ((u >> 16) & 1u);   // round-to-nearest-even
    return (unsigned short)(u >> 16);
}
__device__ __forceinline__ float sigmoidf_(float x) {
    return 1.0f / (1.0f + __expf(-x));
}

// A fragment (16x32 bf16) from row-major bf16 [rows x ldk] at (row0, k0).
// Layout: lanes 0-15 hold K {0..15}; lanes 16-31 hold K {16..31} (matches packed B).
__device__ __forceinline__ v16bf load_a_frag(const unsigned short* A, int ldk,
                                             int row0, int k0) {
    int lane = threadIdx.x & 31;
    int m = lane & 15;
    int g = lane >> 4;
    const unsigned short* p = A + (size_t)(row0 + m) * (size_t)ldk + k0 + g * 16;
    v8bf lo = *reinterpret_cast<const v8bf*>(p);        // K = k0+g*16 .. +7
    v8bf hi = *reinterpret_cast<const v8bf*>(p + 8);    // K = k0+g*16+8 .. +7
    return __builtin_shufflevector(lo, hi, 0, 1, 2, 3, 4, 5, 6, 7,
                                   8, 9, 10, 11, 12, 13, 14, 15);
}

// B fragment from pre-packed buffer: tile stride 512 elements, lane-contiguous.
__device__ __forceinline__ v16bf load_b_frag(const unsigned short* Bp, size_t tile) {
    int lane = threadIdx.x & 31;
    return *reinterpret_cast<const v16bf*>(Bp + tile * 512 + (size_t)lane * 16);
}

__device__ __forceinline__ v8f wmma_bf16(v16bf a, v16bf b, v8f c) {
    return __builtin_amdgcn_wmma_f32_16x16x32_bf16(false, a, false, b,
                                                   (short)0, c, false, false);
}

// ---------- elementwise utility kernels ----------
__global__ void cast_bf16_kernel(const float* __restrict__ src,
                                 unsigned short* __restrict__ dst, int n) {
    int i = blockIdx.x * blockDim.x + threadIdx.x;
    if (i < n) dst[i] = f2bf(src[i]);
}

__global__ void zero_u32_kernel(unsigned int* __restrict__ p, int n) {
    int i = blockIdx.x * blockDim.x + threadIdx.x;
    if (i < n) p[i] = 0u;
}

// Pack row-major fp32 W[K x N] into WMMA B-fragment layout (bf16).
// Tile (kt, nt): base = (nt*Kt + kt)*512; lane holds 16 contiguous bf16:
// element i -> W[kt*32 + (lane>>4)*16 + i][nt*16 + (lane&15)]
__global__ void pack_b_kernel(const float* __restrict__ W,
                              unsigned short* __restrict__ Wp, int K, int N) {
    int lane = threadIdx.x & 31;
    int sub  = threadIdx.x >> 5;
    int nt = blockIdx.x;
    int kt = blockIdx.y * 8 + sub;
    int Kt = K >> 5;
    if (kt >= Kt) return;
    int n = nt * 16 + (lane & 15);
    int kbase = kt * 32 + ((lane >> 4) << 4);
    unsigned short* out = Wp + ((size_t)(nt * Kt + kt) * 512) + (size_t)lane * 16;
#pragma unroll
    for (int i = 0; i < 16; ++i)
        out[i] = f2bf(W[(size_t)(kbase + i) * (size_t)N + n]);
}

// ---------- generic WMMA GEMM: C[MxN] = A_bf16[MxK] @ Bpacked + bias ----------
// block = 256 threads (8 waves). wave -> row tile mt = by*8+wave;
// each wave computes 4 N-tiles (A-fragment reuse x4).
// Software-pipelined: loads for kt+1 are issued before the WMMAs for kt, so
// global-load latency overlaps the matrix pipe instead of s_wait_loadcnt 0.
__global__ void wmma_gemm_bias_kernel(const unsigned short* __restrict__ A,
                                      const unsigned short* __restrict__ Bp,
                                      const float* __restrict__ bias,
                                      float* __restrict__ C,
                                      int M, int N, int K) {
    int wave = threadIdx.x >> 5;
    int lane = threadIdx.x & 31;
    int mt  = blockIdx.y * 8 + wave;
    int nt0 = blockIdx.x * 4;
    int Kt = K >> 5;

    v8f z = {0.f, 0.f, 0.f, 0.f, 0.f, 0.f, 0.f, 0.f};
    v8f acc[4] = {z, z, z, z};

    // prologue: fragments for kt = 0
    v16bf a_cur = load_a_frag(A, K, mt * 16, 0);
    v16bf b_cur[4];
#pragma unroll
    for (int j = 0; j < 4; ++j)
        b_cur[j] = load_b_frag(Bp, (size_t)((nt0 + j) * Kt));

    for (int kt = 0; kt < Kt - 1; ++kt) {
        // issue next k-step loads first (stay outstanding across the WMMAs)
        v16bf a_nxt = load_a_frag(A, K, mt * 16, (kt + 1) * 32);
        v16bf b_nxt[4];
#pragma unroll
        for (int j = 0; j < 4; ++j)
            b_nxt[j] = load_b_frag(Bp, (size_t)((nt0 + j) * Kt + kt + 1));
        if (kt + 2 < Kt) {
            __builtin_prefetch(Bp + ((size_t)(nt0 * Kt + kt + 2)) * 512 +
                               (size_t)lane * 16, 0, 1);
        }
#pragma unroll
        for (int j = 0; j < 4; ++j)
            acc[j] = wmma_bf16(a_cur, b_cur[j], acc[j]);
        a_cur = a_nxt;
#pragma unroll
        for (int j = 0; j < 4; ++j)
            b_cur[j] = b_nxt[j];
    }
    // epilogue: last k-step
#pragma unroll
    for (int j = 0; j < 4; ++j)
        acc[j] = wmma_bf16(a_cur, b_cur[j], acc[j]);

    int g = lane >> 4, n = lane & 15;
#pragma unroll
    for (int j = 0; j < 4; ++j) {
        int col = (nt0 + j) * 16 + n;
        float bv = bias[col];
#pragma unroll
        for (int i = 0; i < 8; ++i) {
            int row = mt * 16 + g * 8 + i;
            C[(size_t)row * (size_t)N + col] = acc[j][i] + bv;
        }
    }
}

// ---------- LSTM step: one launch per timestep ----------
// grid.x = H/16 = 64 (col tile jt); block = 128 (4 waves = 4 row tiles of B=64).
// Each wave computes gate tiles r/f/g/o at columns {0,H,2H,3H}+jt*16, then the
// cell/hidden update in-register. h ping-pongs between two bf16 buffers.
// Same double-buffered pipeline as the generic GEMM.
__global__ void lstm_step_kernel(const unsigned short* __restrict__ h_in,   // [B_ x H_] bf16
                                 const unsigned short* __restrict__ Whp,    // packed [H_ x 4H]
                                 const float* __restrict__ xg,              // [B_,S_,4H]
                                 const float* __restrict__ b_h,             // [4H]
                                 float* __restrict__ c,                     // [B_ x H_]
                                 float* __restrict__ outputs,               // [B_,S_,H_]
                                 unsigned short* __restrict__ outputs_bf,   // [B_,S_,H_]
                                 unsigned short* __restrict__ h_out,        // [B_ x H_] bf16
                                 int t) {
    int wave = threadIdx.x >> 5;   // row tile (batch/16)
    int lane = threadIdx.x & 31;
    int jt = blockIdx.x;           // hidden col tile
    const int Kt = H_ >> 5;        // 32
    const int NT = H_ / 16;        // 64 col tiles per gate

    v8f z = {0.f, 0.f, 0.f, 0.f, 0.f, 0.f, 0.f, 0.f};
    v8f acc[4] = {z, z, z, z};

    v16bf a_cur = load_a_frag(h_in, H_, wave * 16, 0);
    v16bf b_cur[4];
#pragma unroll
    for (int g = 0; g < 4; ++g)
        b_cur[g] = load_b_frag(Whp, (size_t)((g * NT + jt) * Kt));

    for (int kt = 0; kt < Kt - 1; ++kt) {
        v16bf a_nxt = load_a_frag(h_in, H_, wave * 16, (kt + 1) * 32);
        v16bf b_nxt[4];
#pragma unroll
        for (int g = 0; g < 4; ++g)
            b_nxt[g] = load_b_frag(Whp, (size_t)((g * NT + jt) * Kt + kt + 1));
        if (kt + 2 < Kt) {
            __builtin_prefetch(Whp + ((size_t)(jt * Kt + kt + 2)) * 512 +
                               (size_t)lane * 16, 0, 1);
        }
#pragma unroll
        for (int g = 0; g < 4; ++g)
            acc[g] = wmma_bf16(a_cur, b_cur[g], acc[g]);
        a_cur = a_nxt;
#pragma unroll
        for (int g = 0; g < 4; ++g)
            b_cur[g] = b_nxt[g];
    }
#pragma unroll
    for (int g = 0; g < 4; ++g)
        acc[g] = wmma_bf16(a_cur, b_cur[g], acc[g]);

    int grp = lane >> 4, n = lane & 15;
    int col = jt * 16 + n;
#pragma unroll
    for (int i = 0; i < 8; ++i) {
        int b = wave * 16 + grp * 8 + i;                 // batch index
        size_t xbase = (size_t)b * ((size_t)S_ * G4_) + (size_t)t * G4_;
        float r  = sigmoidf_(acc[0][i] + xg[xbase + 0 * H_ + col] + b_h[0 * H_ + col]);
        float f  = sigmoidf_(acc[1][i] + xg[xbase + 1 * H_ + col] + b_h[1 * H_ + col]);
        float gg = tanhf    (acc[2][i] + xg[xbase + 2 * H_ + col] + b_h[2 * H_ + col]);
        float o  = sigmoidf_(acc[3][i] + xg[xbase + 3 * H_ + col] + b_h[3 * H_ + col]);
        size_t cidx = (size_t)b * H_ + col;
        float cn = f * c[cidx] + r * gg;
        c[cidx] = cn;
        float h = o * tanhf(cn);
        size_t oidx = (size_t)b * ((size_t)S_ * H_) + (size_t)t * H_ + col;
        outputs[oidx] = h;
        unsigned short hb = f2bf(h);
        outputs_bf[oidx] = hb;
        h_out[cidx] = hb;
    }
}

// ---------- tail kernels (tiny, scalar fp32) ----------
__global__ void final_hidden_kernel(const float* __restrict__ outputs,
                                    const float* __restrict__ W_lo,
                                    const float* __restrict__ b_lo,
                                    float* __restrict__ fh) {
    int idx = blockIdx.x * blockDim.x + threadIdx.x;   // B_*HD_
    if (idx >= B_ * HD_) return;
    int b = idx >> 9, j = idx & (HD_ - 1);
    const float* hrow = outputs + (size_t)b * ((size_t)S_ * H_) + (size_t)(S_ - 1) * H_;
    float acc = b_lo[j];
    for (int k = 0; k < H_; ++k) acc += hrow[k] * W_lo[(size_t)k * HD_ + j];
    fh[idx] = acc;
}

__global__ void ws_kernel(const float* __restrict__ fh,
                          const float* __restrict__ W_as,
                          const float* __restrict__ b_as,
                          float* __restrict__ WS) {
    int idx = blockIdx.x * blockDim.x + threadIdx.x;   // B_*V_
    if (idx >= B_ * V_) return;
    int b = idx >> 8, v = idx & (V_ - 1);
    float acc = b_as[v];
    for (int k = 0; k < HD_; ++k) acc += fh[(size_t)b * HD_ + k] * W_as[(size_t)k * V_ + v];
    WS[idx] = acc;
}

__global__ void score_kernel(const float* __restrict__ WH,
                             const float* __restrict__ WS,
                             const float* __restrict__ W_v,
                             const float* __restrict__ b_v,
                             float* __restrict__ score) {
    int idx = blockIdx.x * blockDim.x + threadIdx.x;   // B_*S_
    if (idx >= B_ * S_) return;
    int b = idx >> 9;
    const float* wh = WH + (size_t)idx * V_;
    const float* ws = WS + (size_t)b * V_;
    float acc = b_v[0];
    for (int v = 0; v < V_; ++v) acc += tanhf(wh[v] + ws[v]) * W_v[v];
    score[idx] = acc;
}

__global__ void softmax_kernel(const float* __restrict__ score,
                               float* __restrict__ attn) {
    __shared__ float sm[S_];
    int b = blockIdx.x, s = threadIdx.x;
    float x = score[(size_t)b * S_ + s];
    sm[s] = x;
    __syncthreads();
    for (int off = S_ / 2; off > 0; off >>= 1) {
        if (s < off) sm[s] = fmaxf(sm[s], sm[s + off]);
        __syncthreads();
    }
    float mx = sm[0];
    __syncthreads();
    float e = __expf(x - mx);
    sm[s] = e;
    __syncthreads();
    for (int off = S_ / 2; off > 0; off >>= 1) {
        if (s < off) sm[s] += sm[s + off];
        __syncthreads();
    }
    attn[(size_t)b * S_ + s] = e / sm[0];
}

__global__ void attn_pool_kernel(const float* __restrict__ outputs,
                                 const float* __restrict__ attn,
                                 float* __restrict__ attn_out) {
    int b = blockIdx.x;
    for (int h = threadIdx.x; h < H_; h += blockDim.x) {
        float acc = 0.f;
        const float* op = outputs + (size_t)b * ((size_t)S_ * H_) + h;
        const float* ap = attn + (size_t)b * S_;
        for (int s = 0; s < S_; ++s) acc += op[(size_t)s * H_] * ap[s];
        attn_out[(size_t)b * H_ + h] = acc;
    }
}

__global__ void head_kernel(const float* __restrict__ fh,
                            const float* __restrict__ attn_out,
                            const float* __restrict__ W_out,
                            const float* __restrict__ b_out,
                            float* __restrict__ out) {
    int idx = threadIdx.x;                 // B_*NOUT_ = 128
    if (idx >= B_ * NOUT_) return;
    int b = idx >> 1, o = idx & 1;
    float acc = b_out[o];
    for (int k = 0; k < HD_; ++k)
        acc += fh[(size_t)b * HD_ + k] * W_out[(size_t)k * NOUT_ + o];
    for (int k = 0; k < H_; ++k)
        acc += attn_out[(size_t)b * H_ + k] * W_out[(size_t)(HD_ + k) * NOUT_ + o];
    out[idx] = sigmoidf_(acc);
}

// ---------- launcher ----------
extern "C" void kernel_launch(void* const* d_in, const int* in_sizes, int n_in,
                              void* d_out, int out_size, void* d_ws, size_t ws_size,
                              hipStream_t stream) {
    const float* text  = (const float*)d_in[0];
    const float* W_in  = (const float*)d_in[1];
    const float* b_in  = (const float*)d_in[2];
    const float* W_h   = (const float*)d_in[3];
    const float* b_h   = (const float*)d_in[4];
    const float* W_ah  = (const float*)d_in[5];
    const float* b_ah  = (const float*)d_in[6];
    const float* W_as  = (const float*)d_in[7];
    const float* b_as  = (const float*)d_in[8];
    const float* W_v   = (const float*)d_in[9];
    const float* b_v   = (const float*)d_in[10];
    const float* W_lo  = (const float*)d_in[11];
    const float* b_lo  = (const float*)d_in[12];
    const float* W_out = (const float*)d_in[13];
    const float* b_out = (const float*)d_in[14];
    float* out = (float*)d_out;

    char* ws = (char*)d_ws;
    size_t off = 0;
    auto carve = [&](size_t bytes) -> char* {
        off = (off + 255) & ~(size_t)255;
        char* p = ws + off;
        off += bytes;
        return p;
    };

    const size_t nX = (size_t)B_ * S_ * E_;              // 25.2M
    unsigned short* Xbf   = (unsigned short*)carve(nX * 2);
    unsigned short* Winp  = (unsigned short*)carve((size_t)E_ * G4_ * 2);
    unsigned short* Whp   = (unsigned short*)carve((size_t)H_ * G4_ * 2);
    unsigned short* Wahp  = (unsigned short*)carve((size_t)H_ * V_ * 2);
    float*          xg    = (float*)carve((size_t)B_ * S_ * G4_ * 4);
    float*          outputs    = (float*)carve((size_t)B_ * S_ * H_ * 4);
    unsigned short* outputs_bf = (unsigned short*)carve((size_t)B_ * S_ * H_ * 2);
    float*          c     = (float*)carve((size_t)B_ * H_ * 4);
    unsigned short* hbf0  = (unsigned short*)carve((size_t)B_ * H_ * 2);
    unsigned short* hbf1  = (unsigned short*)carve((size_t)B_ * H_ * 2);
    float*          WH    = (float*)carve((size_t)B_ * S_ * V_ * 4);
    float*          fh    = (float*)carve((size_t)B_ * HD_ * 4);
    float*          WS    = (float*)carve((size_t)B_ * V_ * 4);
    float*          score = (float*)carve((size_t)B_ * S_ * 4);
    float*          attn  = (float*)carve((size_t)B_ * S_ * 4);
    float*          attn_out = (float*)carve((size_t)B_ * H_ * 4);
    (void)in_sizes; (void)n_in; (void)out_size; (void)ws_size;

    // 1) activations -> bf16
    cast_bf16_kernel<<<(int)(nX / 256), 256, 0, stream>>>(text, Xbf, (int)nX);

    // 2) pack weights into B-fragment layout
    pack_b_kernel<<<dim3(G4_ / 16, 3), 256, 0, stream>>>(W_in, Winp, E_, G4_);
    pack_b_kernel<<<dim3(G4_ / 16, 4), 256, 0, stream>>>(W_h,  Whp,  H_, G4_);
    pack_b_kernel<<<dim3(V_  / 16, 4), 256, 0, stream>>>(W_ah, Wahp, H_, V_);

    // 3) zero LSTM state (must happen every call: graph replays)
    zero_u32_kernel<<<(B_ * H_) / 256, 256, 0, stream>>>((unsigned int*)c, B_ * H_);
    zero_u32_kernel<<<(B_ * H_ / 2) / 256, 256, 0, stream>>>((unsigned int*)hbf0, B_ * H_ / 2);

    // 4) big fused input GEMM: xg = X @ W_in + b_in   [32768 x 4096]
    wmma_gemm_bias_kernel<<<dim3(G4_ / 64, (B_ * S_) / 128), 256, 0, stream>>>(
        Xbf, Winp, b_in, xg, B_ * S_, G4_, E_);

    // 5) LSTM recurrence, one launch per step, h ping-pong
    for (int t = 0; t < S_; ++t) {
        const unsigned short* hin = (t & 1) ? hbf1 : hbf0;
        unsigned short*       hot = (t & 1) ? hbf0 : hbf1;
        lstm_step_kernel<<<H_ / 16, 128, 0, stream>>>(
            hin, Whp, xg, b_h, c, outputs, outputs_bf, hot, t);
    }

    // 6) attention projection: WH = outputs @ W_ah + b_ah  [32768 x 256]
    wmma_gemm_bias_kernel<<<dim3(V_ / 64, (B_ * S_) / 128), 256, 0, stream>>>(
        outputs_bf, Wahp, b_ah, WH, B_ * S_, V_, H_);

    // 7) tail
    final_hidden_kernel<<<(B_ * HD_) / 256, 256, 0, stream>>>(outputs, W_lo, b_lo, fh);
    ws_kernel<<<(B_ * V_) / 256, 256, 0, stream>>>(fh, W_as, b_as, WS);
    score_kernel<<<(B_ * S_) / 256, 256, 0, stream>>>(WH, WS, W_v, b_v, score);
    softmax_kernel<<<B_, S_, 0, stream>>>(score, attn);
    attn_pool_kernel<<<B_, 256, 0, stream>>>(outputs, attn, attn_out);
    head_kernel<<<1, 128, 0, stream>>>(fh, attn_out, W_out, b_out, out);
}
